// Net_62173946577413
// MI455X (gfx1250) — matrix operand order repaired and checked by reading, not compiled, not run on hardware
//
#include <hip/hip_runtime.h>
#include <hip/hip_bf16.h>

#define F_IN 1433
#define KMAIN 1408     // 44 full 32-wide K chunks
#define KPAD 1440      // F_IN padded to a multiple of 32
#define HID  16
#define NCLS 7

typedef __attribute__((ext_vector_type(16))) __bf16 v16bf;
typedef __attribute__((ext_vector_type(2)))  __bf16 v2bf;
typedef __attribute__((ext_vector_type(8)))  float  v8f;

union AFrag { unsigned u[8]; v16bf v; };
union BFrag { uint4 q[2];   v16bf v; };

// Pack two f32 into one dword of bf16 -- maps to a single v_cvt_pk_bf16_f32.
__device__ __forceinline__ unsigned pack_bf16_pair(float f0, float f1) {
#if __has_builtin(__builtin_amdgcn_cvt_pk_bf16_f32)
  auto p = __builtin_amdgcn_cvt_pk_bf16_f32(f0, f1);
  return __builtin_bit_cast(unsigned, p);
#else
  v2bf p;
  p[0] = (__bf16)f0;
  p[1] = (__bf16)f1;
  return __builtin_bit_cast(unsigned, p);
#endif
}

// ---------------------------------------------------------------------------
// Pack weights: f32 [K=1433][N=16] (row-major) -> bf16 [N=16][KPAD=1440]
// (n-major, K zero-padded) so each lane's B fragment is 2 contiguous uint4.
// ---------------------------------------------------------------------------
__global__ void pack_weights(const float* __restrict__ w_rel1,
                             const float* __restrict__ w_root1,
                             __bf16* __restrict__ wt_rel,
                             __bf16* __restrict__ wt_root) {
  int t = blockIdx.x * blockDim.x + threadIdx.x;
  const int per = HID * KPAD;
  if (t >= 2 * per) return;
  int sel = t / per;
  int r   = t - sel * per;
  int n   = r / KPAD;
  int k   = r - n * KPAD;
  const float* w = sel ? w_root1 : w_rel1;
  __bf16 v = (k < F_IN) ? (__bf16)w[k * HID + n] : (__bf16)0.0f;
  (sel ? wt_root : wt_rel)[n * KPAD + k] = v;
}

// ---------------------------------------------------------------------------
// Layer-1 fused GEMM: hrel = x @ w_rel1, troot = x @ w_root1.
// One wave per 16-node tile; A fragment shared by two WMMA accumulators.
// Per-wave LDS staging; CDNA5 LDS ops are in-order within a wave, so only a
// compiler scheduling fence (wave_barrier) is needed -- no s_barrier.
// ---------------------------------------------------------------------------
__global__ void __launch_bounds__(128)
gemm1_wmma(const float* __restrict__ x,
           const __bf16* __restrict__ wt_rel,
           const __bf16* __restrict__ wt_root,
           float* __restrict__ hrel,
           float* __restrict__ troot,
           int nNodes) {
  __shared__ __bf16 tileA[4][16 * 32];   // per-wave 16x32 bf16 staging (4 KB)

  const int lane = threadIdx.x & 31;
  const int wave = threadIdx.x >> 5;

  const int numTiles = nNodes >> 4;               // N is a multiple of 16
  int tile = blockIdx.x * 4 + wave;
  if (tile >= numTiles) tile = numTiles - 1;      // keep EXEC all-1s for WMMA
  const int rowBase = tile << 4;

  // staging role: 2 lanes per row, 16 contiguous K floats each (coalesced)
  const int rStage = lane >> 1;
  const int cStage = (lane & 1) << 4;
  const float* xr = x + (size_t)(rowBase + rStage) * F_IN + cStage;
  unsigned* stp = (unsigned*)&tileA[wave][rStage * 32 + cStage];

  // WMMA fragment roles (16-bit A 16x32 / B 32x16, both k = 16*(lane/16) + s, wave32)
  const int kOffA = (lane & 16) ? 16 : 0;
  const int kOffB = (lane & 16) ? 16 : 0;
  const int n     = lane & 15;
  const unsigned* ldsRow = (const unsigned*)&tileA[wave][(lane & 15) * 32];
  const __bf16* wrelBase  = wt_rel  + (size_t)n * KPAD + kOffB;
  const __bf16* wrootBase = wt_root + (size_t)n * KPAD + kOffB;

  v8f accRel  = {};
  v8f accRoot = {};

  // ---------------- main loop: 44 full 32-wide chunks, no bounds checks ----
  for (int kb = 0; kb < KMAIN; kb += 32) {
    __builtin_prefetch(xr + kb + 32, 0, 1);       // global_prefetch_b8

#pragma unroll
    for (int j = 0; j < 8; ++j)                   // 16 floats -> 8 cvt_pk
      stp[j] = pack_bf16_pair(xr[kb + 2 * j], xr[kb + 2 * j + 1]);

    __builtin_amdgcn_wave_barrier();              // LDS in-order within wave

    AFrag a;
#pragma unroll
    for (int vi = 0; vi < 8; ++vi) {
      int k = kOffA + 2 * vi;
      a.u[vi] = ldsRow[k >> 1];
    }

    BFrag brel, broot;
    const uint4* bq  = (const uint4*)(wrelBase  + kb);
    const uint4* bq2 = (const uint4*)(wrootBase + kb);
    brel.q[0]  = bq[0];  brel.q[1]  = bq[1];
    broot.q[0] = bq2[0]; broot.q[1] = bq2[1];

    accRel = __builtin_amdgcn_wmma_f32_16x16x32_bf16(
        false, a.v, false, brel.v, (short)0, accRel, false, false);
    accRoot = __builtin_amdgcn_wmma_f32_16x16x32_bf16(
        false, a.v, false, broot.v, (short)0, accRoot, false, false);

    __builtin_amdgcn_wave_barrier();              // WAR fence before restage
  }

  // ---------------- tail chunk (K = 1408..1432, zero-padded to 1440) -------
  {
    const int kb = KMAIN;
#pragma unroll
    for (int j = 0; j < 8; ++j) {
      int k0 = kb + cStage + 2 * j;
      float f0 = (k0     < F_IN) ? xr[kb + 2 * j]     : 0.0f;
      float f1 = (k0 + 1 < F_IN) ? xr[kb + 2 * j + 1] : 0.0f;
      stp[j] = pack_bf16_pair(f0, f1);
    }
    __builtin_amdgcn_wave_barrier();

    AFrag a;
#pragma unroll
    for (int vi = 0; vi < 8; ++vi) {
      int k = kOffA + 2 * vi;
      a.u[vi] = ldsRow[k >> 1];
    }

    BFrag brel, broot;
    const uint4* bq  = (const uint4*)(wrelBase  + kb);
    const uint4* bq2 = (const uint4*)(wrootBase + kb);
    brel.q[0]  = bq[0];  brel.q[1]  = bq[1];
    broot.q[0] = bq2[0]; broot.q[1] = bq2[1];

    accRel = __builtin_amdgcn_wmma_f32_16x16x32_bf16(
        false, a.v, false, brel.v, (short)0, accRel, false, false);
    accRoot = __builtin_amdgcn_wmma_f32_16x16x32_bf16(
        false, a.v, false, broot.v, (short)0, accRoot, false, false);
  }

  // ---- store D: VGPR r -> M = r (lanes 0-15) / r+8 (lanes 16-31), N = lane&15
  const int halfRow = (lane & 16) ? 8 : 0;
#pragma unroll
  for (int r = 0; r < 8; ++r) {
    size_t row = (size_t)(rowBase + r + halfRow);
    hrel [row * HID + n] = accRel[r];
    troot[row * HID + n] = accRoot[r];
  }
}

// ---------------------------------------------------------------------------
// Edge scatter layer 1: agg[dst] += hrel[src]  (16 channels, 4 per thread)
// ---------------------------------------------------------------------------
__global__ void scatter1(const int* __restrict__ src, const int* __restrict__ dst,
                         const float* __restrict__ hrel, float* __restrict__ agg,
                         int e4) {
  int t = blockIdx.x * blockDim.x + threadIdx.x;
  if (t >= e4) return;
  int e = t >> 2;
  int q = (t & 3) << 2;
  int s = src[e], d = dst[e];
  const float4 v = *(const float4*)(hrel + (size_t)s * HID + q);
  float* a = agg + (size_t)d * HID + q;
  atomicAdd(a + 0, v.x);
  atomicAdd(a + 1, v.y);
  atomicAdd(a + 2, v.z);
  atomicAdd(a + 3, v.w);
}

// ---------------------------------------------------------------------------
// h1 = elu(agg + b_rel1 + troot);  hrel2 = h1 @ w_rel2  (padded to 8 cols)
// ---------------------------------------------------------------------------
__global__ void l1_finish(const float* __restrict__ agg,
                          const float* __restrict__ troot,
                          const float* __restrict__ b_rel1,
                          const float* __restrict__ w_rel2,
                          float* __restrict__ h1,
                          float* __restrict__ hrel2,
                          int nNodes) {
  __shared__ float wsh[HID * NCLS];
  __shared__ float bsh[HID];
  if (threadIdx.x < HID * NCLS) wsh[threadIdx.x] = w_rel2[threadIdx.x];
  if (threadIdx.x < HID)        bsh[threadIdx.x] = b_rel1[threadIdx.x];
  __syncthreads();

  int nIdx = blockIdx.x * blockDim.x + threadIdx.x;
  if (nIdx >= nNodes) return;

  float h[HID];
#pragma unroll
  for (int k = 0; k < HID; ++k) {
    float v = agg[(size_t)nIdx * HID + k] + bsh[k] + troot[(size_t)nIdx * HID + k];
    h[k] = (v > 0.0f) ? v : (expf(v) - 1.0f);   // elu
    h1[(size_t)nIdx * HID + k] = h[k];
  }
#pragma unroll
  for (int c = 0; c < NCLS; ++c) {
    float s = 0.0f;
#pragma unroll
    for (int k = 0; k < HID; ++k) s += h[k] * wsh[k * NCLS + c];
    hrel2[(size_t)nIdx * 8 + c] = s;
  }
  hrel2[(size_t)nIdx * 8 + 7] = 0.0f;
}

// ---------------------------------------------------------------------------
// Edge scatter layer 2: agg2[dst][0..6] += hrel2[src][0..6]
// ---------------------------------------------------------------------------
__global__ void scatter2(const int* __restrict__ src, const int* __restrict__ dst,
                         const float* __restrict__ hrel2, float* __restrict__ agg2,
                         int nEdges) {
  int e = blockIdx.x * blockDim.x + threadIdx.x;
  if (e >= nEdges) return;
  int s = src[e], d = dst[e];
  const float4 v0 = *(const float4*)(hrel2 + (size_t)s * 8);
  const float4 v1 = *(const float4*)(hrel2 + (size_t)s * 8 + 4);
  float* a = agg2 + (size_t)d * 8;
  atomicAdd(a + 0, v0.x);
  atomicAdd(a + 1, v0.y);
  atomicAdd(a + 2, v0.z);
  atomicAdd(a + 3, v0.w);
  atomicAdd(a + 4, v1.x);
  atomicAdd(a + 5, v1.y);
  atomicAdd(a + 6, v1.z);
}

// ---------------------------------------------------------------------------
// out = log_softmax(agg2 + b_rel2 + h1 @ w_root2)
// ---------------------------------------------------------------------------
__global__ void finalize(const float* __restrict__ agg2,
                         const float* __restrict__ h1,
                         const float* __restrict__ w_root2,
                         const float* __restrict__ b_rel2,
                         float* __restrict__ out,
                         int nNodes) {
  __shared__ float wsh[HID * NCLS];
  __shared__ float bsh[NCLS];
  if (threadIdx.x < HID * NCLS) wsh[threadIdx.x] = w_root2[threadIdx.x];
  if (threadIdx.x < NCLS)       bsh[threadIdx.x] = b_rel2[threadIdx.x];
  __syncthreads();

  int nIdx = blockIdx.x * blockDim.x + threadIdx.x;
  if (nIdx >= nNodes) return;

  float h[HID];
#pragma unroll
  for (int k = 0; k < HID; ++k) h[k] = h1[(size_t)nIdx * HID + k];

  float o[NCLS];
  float mx = -3.4e38f;
#pragma unroll
  for (int c = 0; c < NCLS; ++c) {
    float s = agg2[(size_t)nIdx * 8 + c] + bsh[c];
#pragma unroll
    for (int k = 0; k < HID; ++k) s += h[k] * wsh[k * NCLS + c];
    o[c] = s;
    mx = fmaxf(mx, s);
  }
  float se = 0.0f;
#pragma unroll
  for (int c = 0; c < NCLS; ++c) se += expf(o[c] - mx);
  float lse = logf(se);
#pragma unroll
  for (int c = 0; c < NCLS; ++c) out[(size_t)nIdx * NCLS + c] = o[c] - mx - lse;
}

// ---------------------------------------------------------------------------
extern "C" void kernel_launch(void* const* d_in, const int* in_sizes, int n_in,
                              void* d_out, int out_size, void* d_ws, size_t ws_size,
                              hipStream_t stream) {
  const float* x       = (const float*)d_in[0];
  const int*   ei      = (const int*)  d_in[1];
  const float* w_root1 = (const float*)d_in[2];
  const float* w_rel1  = (const float*)d_in[3];
  const float* b_rel1  = (const float*)d_in[4];
  const float* w_root2 = (const float*)d_in[5];
  const float* w_rel2  = (const float*)d_in[6];
  const float* b_rel2  = (const float*)d_in[7];

  const int N = in_sizes[0] / F_IN;    // 100000
  const int E = in_sizes[1] / 2;       // 3200000
  const int* srcI = ei;
  const int* dstI = ei + E;

  // workspace layout (all 256B aligned)
  char* ws = (char*)d_ws;
  size_t off = 0;
  __bf16* wt_rel  = (__bf16*)(ws + off); off += (size_t)HID * KPAD * 2;  // 46080
  __bf16* wt_root = (__bf16*)(ws + off); off += (size_t)HID * KPAD * 2;
  float*  hrel    = (float*)(ws + off);  off += (size_t)N * HID * 4;
  float*  troot   = (float*)(ws + off);  off += (size_t)N * HID * 4;
  float*  agg     = (float*)(ws + off);  off += (size_t)N * HID * 4;
  float*  h1      = (float*)(ws + off);  off += (size_t)N * HID * 4;
  float*  hrel2   = (float*)(ws + off);  off += (size_t)N * 8 * 4;
  float*  agg2    = (float*)(ws + off);  off += (size_t)N * 8 * 4;

  // zero the scatter accumulators every call (graph-capture-safe memsets)
  hipMemsetAsync(agg,  0, (size_t)N * HID * 4, stream);
  hipMemsetAsync(agg2, 0, (size_t)N * 8 * 4,   stream);

  // 1) pack weights to bf16 n-major, K padded
  {
    int total = 2 * HID * KPAD;
    pack_weights<<<(total + 255) / 256, 256, 0, stream>>>(w_rel1, w_root1,
                                                          wt_rel, wt_root);
  }
  // 2) layer-1 fused WMMA GEMM
  {
    int numTiles = N >> 4;                       // 6250
    int blocks = (numTiles + 3) / 4;
    gemm1_wmma<<<blocks, 128, 0, stream>>>(x, wt_rel, wt_root, hrel, troot, N);
  }
  // 3) layer-1 edge scatter
  {
    int e4 = E * 4;
    scatter1<<<(e4 + 255) / 256, 256, 0, stream>>>(srcI, dstI, hrel, agg, e4);
  }
  // 4) ELU + layer-2 rel projection
  l1_finish<<<(N + 255) / 256, 256, 0, stream>>>(agg, troot, b_rel1, w_rel2,
                                                 h1, hrel2, N);
  // 5) layer-2 edge scatter
  scatter2<<<(E + 255) / 256, 256, 0, stream>>>(srcI, dstI, hrel2, agg2, E);
  // 6) root2 + bias + log_softmax
  finalize<<<(N + 255) / 256, 256, 0, stream>>>(agg2, h1, w_root2, b_rel2,
                                                (float*)d_out, N);
}
